// GPT_32959579029946
// MI455X (gfx1250) — matrix operand order repaired and checked by reading, not compiled, not run on hardware
//
#include <hip/hip_runtime.h>
#include <hip/hip_bf16.h>

// Tiny GPT forward, fully fused. One wave32 = one sequence.
// All GEMMs on v_wmma_f32_16x16x32_f16; weights pre-tiled to f16 WMMA
// B-fragment order in d_ws (one aligned 32B load per lane per fragment).
// A-fragments hoisted and reused across all N-tiles of each GEMM.

typedef __attribute__((ext_vector_type(16))) _Float16 v16h;
typedef __attribute__((ext_vector_type(8)))  _Float16 v8h;
typedef __attribute__((ext_vector_type(8)))  float    v8f;

#define V_  65
#define E_  64
#define T_  32
#define H_  4
#define HS_ 16
#define L_  2
#define FF_ 256
#define B_  8192
#define WAVES 4

// fragment-buffer offsets in d_ws (units: _Float16). frag = 512 halves.
// frag index within a matrix: ((l*NT + nt)*KT + kt)
#define OFF_WQ 0            // L*4*2 frags  = 16 frags = 8192 halves
#define OFF_WK 8192
#define OFF_WV 16384
#define OFF_WO 24576
#define OFF_W1 32768        // L*16*2 frags = 64 frags = 32768 halves
#define OFF_W2 65536        // L*4*8 frags  = 64 frags = 32768 halves
#define OFF_WH 98304        // 5*2 frags    = 10 frags = 5120 halves

static __device__ __forceinline__ void wave_fence() {
  asm volatile("s_wait_dscnt 0x0" ::: "memory");
}

static __device__ __forceinline__ v8f wmma_f16(v16h a, v16h b, v8f c) {
  return __builtin_amdgcn_wmma_f32_16x16x32_f16(
      false, a, false, b, (short)0, c, false, false);
}

// ---- weight pre-tiling: f32 [L][K][N] -> f16 B-fragments ------------------
// B 32x16 frag layout used here (contiguous k, as load_A16): lane n<16 -> col n, K=k0..k0+15;
// lane n+16 -> col n, K=k0+16..k0+31; 16 contiguous halves per lane.
__global__ void prep_w(const float* __restrict__ W, _Float16* __restrict__ out,
                       int K, int N, int KT, int NT, int Lc) {
  const int tid   = blockIdx.x * blockDim.x + threadIdx.x;
  const int total = Lc * KT * NT * 512;
  if (tid >= total) return;
  const int j    = tid & 15;
  const int lane = (tid >> 4) & 31;
  const int f    = tid >> 9;
  const int kt   = f % KT;
  const int nt   = (f / KT) % NT;
  const int l    = f / (KT * NT);
  const int col  = nt * 16 + (lane & 15);
  const int k    = kt * 32 + ((lane & 16) ? 16 : 0) + j;
  const float v  = (col < N) ? W[(size_t)l * K * N + (size_t)k * N + col] : 0.f;
  out[tid] = (_Float16)v;
}

// ---- fragment helpers (wave32; A/B use contiguous k, not ISA table order) ---------------

// B-fragment: one aligned 32B load per lane.
static __device__ __forceinline__ v16h load_F(const _Float16* F, int lane) {
  return *(const v16h*)(F + lane * 16);
}

// A 16x32 f16 from row-major LDS: lanes 0-15 row M=lane, K=k0..k0+15;
// lanes 16-31 row M=lane-16, K=k0+16..k0+31.
static __device__ __forceinline__ v16h load_A16(const _Float16* A, int lda,
                                                int m0, int k0, int lane) {
  const int row = m0 + (lane & 15);
  const int kb  = k0 + ((lane & 16) ? 16 : 0);
  const _Float16* p = A + row * lda + kb;
  v16h a;
#pragma unroll
  for (int j = 0; j < 8; ++j) a[j] = p[j];
#pragma unroll
  for (int j = 0; j < 8; ++j) a[8 + j] = p[8 + j];
  return a;
}

// D 16x16 f32: lanes 0-15 rows m0..m0+7 (VGPR r = row r); lanes 16-31 rows +8.
static __device__ __forceinline__ void store_D_f32(float* D, int ldd, int m0,
                                                   int n0, v8f d, int lane) {
  const int col = n0 + (lane & 15);
  const int r0  = m0 + ((lane & 16) ? 8 : 0);
#pragma unroll
  for (int r = 0; r < 8; ++r) D[(r0 + r) * ldd + col] = d[r];
}

static __device__ __forceinline__ void store_D_f16(_Float16* D, int ldd, int m0,
                                                   int n0, v8f d, int lane) {
  const int col = n0 + (lane & 15);
  const int r0  = m0 + ((lane & 16) ? 8 : 0);
#pragma unroll
  for (int r = 0; r < 8; ++r) D[(r0 + r) * ldd + col] = (_Float16)d[r];
}

// K-proj D tile (token rows, head-dims cols) -> score B-fragment.
// B[k=d][n=t]: k<16 lives in frag lanes 0-15 (lane = t&15, elem j = d);
// k>=16 (zero pad) lives in lanes 16-31 (pre-zeroed).
static __device__ __forceinline__ void store_Kfrag(_Float16* F, v8f d,
                                                   int lane) {
  const int r0 = (lane & 16) ? 8 : 0;
  const int c  = lane & 15;
#pragma unroll
  for (int r = 0; r < 8; ++r) F[(r0 + r) * 16 + c] = (_Float16)d[r];
}

// V-proj D tile (token rows t, head-dim cols d) -> PV B-fragment.
// B[k=t][n=d]: frag lane = (d&15) + (t>=16 ? 16 : 0), elem j = t&15.
static __device__ __forceinline__ void store_Vfrag(_Float16* F, int mt, v8f d,
                                                   int lane) {
  const int r0 = (lane & 16) ? 8 : 0;
  const int Lb = (lane & 15) + mt * 16;
#pragma unroll
  for (int r = 0; r < 8; ++r) F[Lb * 16 + (r0 + r)] = (_Float16)d[r];
}

static __device__ __forceinline__ void add_D_bias(float* X, int ldd, int m0,
                                                  int n0, v8f d,
                                                  const float* bias, int lane) {
  const int col = n0 + (lane & 15);
  const int r0  = m0 + ((lane & 16) ? 8 : 0);
  const float bv = bias[col];
#pragma unroll
  for (int r = 0; r < 8; ++r) X[(r0 + r) * ldd + col] += d[r] + bv;
}

static __device__ __forceinline__ void store_D_relu_f16(_Float16* D, int ldd,
                                                        int m0, int n0, v8f d,
                                                        const float* bias,
                                                        int lane) {
  const int col = n0 + (lane & 15);
  const int r0  = m0 + ((lane & 16) ? 8 : 0);
  const float bv = bias[col];
#pragma unroll
  for (int r = 0; r < 8; ++r) {
    float x = d[r] + bv;
    D[(r0 + r) * ldd + col] = (_Float16)(x > 0.f ? x : 0.f);
  }
}

// Row LayerNorm: lane t owns row t (T_ == wave size).
static __device__ __forceinline__ void ln_rows(const float (*X)[E_],
                                               _Float16 (*O)[E_],
                                               const float* g, const float* b,
                                               int lane) {
  const int t = lane;
  float m = 0.f;
  for (int e = 0; e < E_; ++e) m += X[t][e];
  m *= (1.f / E_);
  float v = 0.f;
  for (int e = 0; e < E_; ++e) { float d = X[t][e] - m; v += d * d; }
  v *= (1.f / E_);
  const float r = rsqrtf(v + 1e-5f);
  for (int e = 0; e < E_; ++e)
    O[t][e] = (_Float16)((X[t][e] - m) * r * g[e] + b[e]);
}

__global__ __launch_bounds__(WAVES * 32) void gpt_fused_fwd(
    const int* __restrict__ idx, const float* __restrict__ tok_emb,
    const float* __restrict__ pos_emb, const _Float16* __restrict__ wf,
    const float* __restrict__ bo, const float* __restrict__ ln1_g,
    const float* __restrict__ ln1_b, const float* __restrict__ ln2_g,
    const float* __restrict__ ln2_b, const float* __restrict__ b1,
    const float* __restrict__ b2, const float* __restrict__ lnf_g,
    const float* __restrict__ lnf_b, const float* __restrict__ bh,
    float* __restrict__ out) {
  __shared__ float    Xr_s[WAVES][T_][E_];   // residual stream (f32)
  __shared__ _Float16 Af_s[WAVES][T_][E_];   // A-operand staging
  __shared__ float    Sf_s[WAVES][T_][T_];   // attention scores
  __shared__ _Float16 Pf_s[WAVES][T_][T_];   // softmax probs (A operand)
  __shared__ __align__(32) unsigned char Scr_s[WAVES][20480];

  const int wave = threadIdx.x >> 5;
  const int lane = threadIdx.x & 31;
  const int seq  = blockIdx.x * WAVES + wave;

  float    (*Xr)[E_] = Xr_s[wave];
  _Float16 (*Af)[E_] = Af_s[wave];
  float    (*Sf)[T_] = Sf_s[wave];
  _Float16 (*Pf)[T_] = Pf_s[wave];
  unsigned char* scr = Scr_s[wave];
  _Float16* Qp  = (_Float16*)scr;             // [H][32][32] row-major, K pad
  _Float16* Ktf = (_Float16*)(scr + 8192);    // [H][2] score B-frags (512 ea)
  _Float16* Vff = (_Float16*)(scr + 16384);   // [H] PV B-frags (512 ea)
  _Float16 (*Hf)[FF_] = (_Float16(*)[FF_])scr;// [32][256] (MLP phase)

  // ---- embedding: lane t owns token t ----
  {
    const int t   = lane;
    const int tok = idx[seq * T_ + t];
    const float* te = tok_emb + (size_t)tok * E_;
    const float* pe = pos_emb + (size_t)t * E_;
#pragma unroll
    for (int e = 0; e < E_; e += 4) {
      float4 a = *(const float4*)(te + e);
      float4 b = *(const float4*)(pe + e);
      float4 r = make_float4(a.x + b.x, a.y + b.y, a.z + b.z, a.w + b.w);
      *(float4*)&Xr[t][e] = r;
    }
  }
  wave_fence();

  const float scale = 0.125f;  // E^-0.5

  for (int l = 0; l < L_; ++l) {
    // ---- LN1 ----
    ln_rows(Xr, Af, ln1_g + l * E_, ln1_b + l * E_, lane);

    // zero Qp (K-pad) + Ktf (k>=16 half) : 8192 contiguous halves
    {
      _Float16* z = (_Float16*)scr;
      const v8h zero = {};
#pragma unroll
      for (int i = lane * 8; i < 8192; i += 256) *(v8h*)(z + i) = zero;
    }
    wave_fence();

    // ---- QKV: [32x64] @ [64x64]; N-tile nt == head ----
    {
      v16h a[2][2];
#pragma unroll
      for (int mt = 0; mt < 2; ++mt)
#pragma unroll
        for (int kt = 0; kt < 2; ++kt)
          a[mt][kt] = load_A16(&Af[0][0], E_, mt * 16, kt * 32, lane);

#pragma unroll
      for (int mt = 0; mt < 2; ++mt) {
#pragma unroll
        for (int nt = 0; nt < H_; ++nt) {
          v8f q = {}, k = {}, v = {};
#pragma unroll
          for (int kt = 0; kt < 2; ++kt) {
            const int f = ((l * 4 + nt) * 2 + kt) << 9;
            v16h bq = load_F(wf + OFF_WQ + f, lane);
            v16h bk = load_F(wf + OFF_WK + f, lane);
            v16h bv = load_F(wf + OFF_WV + f, lane);
            q = wmma_f16(a[mt][kt], bq, q);
            k = wmma_f16(a[mt][kt], bk, k);
            v = wmma_f16(a[mt][kt], bv, v);
          }
          store_D_f16(Qp + nt * 1024, 32, mt * 16, 0, q, lane);   // Q[t][d]
          store_Kfrag(Ktf + (nt * 2 + mt) * 512, k, lane);        // score frag
          store_Vfrag(Vff + nt * 512, mt, v, lane);               // PV frag
        }
      }
    }
    wave_fence();

    // ---- attention per head; concat outputs into Af ----
    for (int h = 0; h < H_; ++h) {
      const _Float16* Qh = Qp + h * 1024;
      {
        v16h aq[2];
#pragma unroll
        for (int mt = 0; mt < 2; ++mt)
          aq[mt] = load_A16(Qh, 32, mt * 16, 0, lane);
#pragma unroll
        for (int mt = 0; mt < 2; ++mt)
#pragma unroll
          for (int nt = 0; nt < 2; ++nt) {
            v8f s = {};
            v16h b = load_F(Ktf + (h * 2 + nt) * 512, lane);
            s = wmma_f16(aq[mt], b, s);
            store_D_f32(&Sf[0][0], T_, mt * 16, nt * 16, s, lane);
          }
      }
      wave_fence();

      // causal softmax: lane t owns row t
      {
        const int t = lane;
        float mx = -3.0e38f;
        for (int s = 0; s <= t; ++s) {
          float x = Sf[t][s] * scale;
          Sf[t][s] = x;
          mx = x > mx ? x : mx;
        }
        float sum = 0.f;
        for (int s = 0; s <= t; ++s) {
          float e = __expf(Sf[t][s] - mx);
          Sf[t][s] = e;
          sum += e;
        }
        const float inv = 1.f / sum;
        for (int s = 0; s < T_; ++s)
          Pf[t][s] = (s <= t) ? (_Float16)(Sf[t][s] * inv) : (_Float16)0.f;
      }
      wave_fence();

      // O_h = P[32x32] @ V_h[32x16] (exact K=32)
      {
        v16h bv = load_F(Vff + h * 512, lane);
#pragma unroll
        for (int mt = 0; mt < 2; ++mt) {
          v8f o = {};
          v16h ap = load_A16(&Pf[0][0], T_, mt * 16, 0, lane);
          o = wmma_f16(ap, bv, o);
          store_D_f16(&Af[0][0], E_, mt * 16, h * 16, o, lane);
        }
      }
      wave_fence();
    }

    // ---- output projection + bias + residual ----
    {
      v16h a[2][2];
#pragma unroll
      for (int mt = 0; mt < 2; ++mt)
#pragma unroll
        for (int kt = 0; kt < 2; ++kt)
          a[mt][kt] = load_A16(&Af[0][0], E_, mt * 16, kt * 32, lane);
#pragma unroll
      for (int mt = 0; mt < 2; ++mt)
#pragma unroll
        for (int nt = 0; nt < 4; ++nt) {
          v8f acc = {};
#pragma unroll
          for (int kt = 0; kt < 2; ++kt) {
            v16h b = load_F(wf + OFF_WO + (((l * 4 + nt) * 2 + kt) << 9), lane);
            acc = wmma_f16(a[mt][kt], b, acc);
          }
          add_D_bias(&Xr[0][0], E_, mt * 16, nt * 16, acc, bo + l * E_, lane);
        }
    }
    wave_fence();

    // ---- LN2 ----
    ln_rows(Xr, Af, ln2_g + l * E_, ln2_b + l * E_, lane);
    wave_fence();

    // ---- MLP up: relu([32x64]@[64x256] + b1) -> Hf ----
    {
      v16h a[2][2];
#pragma unroll
      for (int mt = 0; mt < 2; ++mt)
#pragma unroll
        for (int kt = 0; kt < 2; ++kt)
          a[mt][kt] = load_A16(&Af[0][0], E_, mt * 16, kt * 32, lane);
#pragma unroll
      for (int mt = 0; mt < 2; ++mt)
        for (int nt = 0; nt < FF_ / 16; ++nt) {
          v8f acc = {};
#pragma unroll
          for (int kt = 0; kt < 2; ++kt) {
            v16h b = load_F(wf + OFF_W1 + (((l * 16 + nt) * 2 + kt) << 9), lane);
            acc = wmma_f16(a[mt][kt], b, acc);
          }
          store_D_relu_f16(&Hf[0][0], FF_, mt * 16, nt * 16, acc, b1 + l * FF_,
                           lane);
        }
    }
    wave_fence();

    // ---- MLP down: [32x256]@[256x64] + b2 + residual ----
#pragma unroll
    for (int mt = 0; mt < 2; ++mt) {
      v16h a[FF_ / 32];
#pragma unroll
      for (int kt = 0; kt < FF_ / 32; ++kt)
        a[kt] = load_A16(&Hf[0][0], FF_, mt * 16, kt * 32, lane);
#pragma unroll
      for (int nt = 0; nt < 4; ++nt) {
        v8f acc = {};
#pragma unroll
        for (int kt = 0; kt < FF_ / 32; ++kt) {
          v16h b = load_F(wf + OFF_W2 + (((l * 4 + nt) * 8 + kt) << 9), lane);
          acc = wmma_f16(a[kt], b, acc);
        }
        add_D_bias(&Xr[0][0], E_, mt * 16, nt * 16, acc, b2 + l * E_, lane);
      }
    }
    wave_fence();
  }

  // ---- final LN + vocab head (V padded to 80, guarded stores) ----
  ln_rows(Xr, Af, lnf_g, lnf_b, lane);
  wave_fence();

  {
    v16h a[2][2];
#pragma unroll
    for (int mt = 0; mt < 2; ++mt)
#pragma unroll
      for (int kt = 0; kt < 2; ++kt)
        a[mt][kt] = load_A16(&Af[0][0], E_, mt * 16, kt * 32, lane);
#pragma unroll
    for (int mt = 0; mt < 2; ++mt)
#pragma unroll
      for (int nt = 0; nt < 5; ++nt) {
        v8f acc = {};
#pragma unroll
        for (int kt = 0; kt < 2; ++kt) {
          v16h b = load_F(wf + OFF_WH + ((nt * 2 + kt) << 9), lane);
          acc = wmma_f16(a[mt][kt], b, acc);
        }
        const int col = nt * 16 + (lane & 15);
        const int r0  = mt * 16 + ((lane & 16) ? 8 : 0);
        if (col < V_) {
          const float bv = bh[col];
#pragma unroll
          for (int r = 0; r < 8; ++r)
            out[(size_t)(seq * T_ + r0 + r) * V_ + col] = acc[r] + bv;
        }
      }
  }
}

extern "C" void kernel_launch(void* const* d_in, const int* in_sizes, int n_in,
                              void* d_out, int out_size, void* d_ws,
                              size_t ws_size, hipStream_t stream) {
  (void)in_sizes; (void)n_in; (void)ws_size; (void)out_size;
  const int*   idx     = (const int*)d_in[0];
  const float* tok_emb = (const float*)d_in[1];
  const float* pos_emb = (const float*)d_in[2];
  const float* Wq      = (const float*)d_in[3];
  const float* Wk      = (const float*)d_in[4];
  const float* Wv      = (const float*)d_in[5];
  const float* Wo      = (const float*)d_in[6];
  const float* bo      = (const float*)d_in[7];
  const float* ln1_g   = (const float*)d_in[8];
  const float* ln1_b   = (const float*)d_in[9];
  const float* ln2_g   = (const float*)d_in[10];
  const float* ln2_b   = (const float*)d_in[11];
  const float* W1      = (const float*)d_in[12];
  const float* b1      = (const float*)d_in[13];
  const float* W2      = (const float*)d_in[14];
  const float* b2      = (const float*)d_in[15];
  const float* lnf_g   = (const float*)d_in[16];
  const float* lnf_b   = (const float*)d_in[17];
  const float* Wh      = (const float*)d_in[18];
  const float* bh      = (const float*)d_in[19];
  float* out = (float*)d_out;
  _Float16* wf = (_Float16*)d_ws;

  // ---- pre-tile weights into WMMA B-fragment order (f16) ----
  auto blocks = [](int n) { return (n + 255) / 256; };
  prep_w<<<blocks(L_ * 2 * 4 * 512), 256, 0, stream>>>(Wq, wf + OFF_WQ, E_, E_, 2, 4, L_);
  prep_w<<<blocks(L_ * 2 * 4 * 512), 256, 0, stream>>>(Wk, wf + OFF_WK, E_, E_, 2, 4, L_);
  prep_w<<<blocks(L_ * 2 * 4 * 512), 256, 0, stream>>>(Wv, wf + OFF_WV, E_, E_, 2, 4, L_);
  prep_w<<<blocks(L_ * 2 * 4 * 512), 256, 0, stream>>>(Wo, wf + OFF_WO, E_, E_, 2, 4, L_);
  prep_w<<<blocks(L_ * 2 * 16 * 512), 256, 0, stream>>>(W1, wf + OFF_W1, E_, FF_, 2, 16, L_);
  prep_w<<<blocks(L_ * 8 * 4 * 512), 256, 0, stream>>>(W2, wf + OFF_W2, FF_, E_, 8, 4, L_);
  prep_w<<<blocks(2 * 5 * 512), 256, 0, stream>>>(Wh, wf + OFF_WH, E_, V_, 2, 5, 1);

  gpt_fused_fwd<<<dim3(B_ / WAVES), dim3(WAVES * 32), 0, stream>>>(
      idx, tok_emb, pos_emb, wf, bo, ln1_g, ln1_b, ln2_g, ln2_b, b1, b2, lnf_g,
      lnf_b, bh, out);
}
